// LlamaAttention_heavy_hitter_our_sketch_17351667876195
// MI455X (gfx1250) — matrix operand order repaired and checked by reading, not compiled, not run on hardware
//
#include <hip/hip_runtime.h>
#include <hip/hip_fp16.h>

typedef unsigned short u16;
typedef __attribute__((ext_vector_type(16))) __bf16    v16bf;
typedef __attribute__((ext_vector_type(16))) _Float16  v16h;
typedef __attribute__((ext_vector_type(8)))  float     v8f;
typedef __attribute__((ext_vector_type(4)))  unsigned int u32x4;
typedef __attribute__((ext_vector_type(8)))  int          i32x8;
typedef __attribute__((ext_vector_type(4)))  int          i32x4;

union Frag { uint4 u[2]; v16bf bf; v16h h; };

#define HIDDEN 2048
#define NHEAD  16
#define DHEAD  128
#define SEQ    2048
#define RREC   409
#define CKEEP  818      // R + HB
#define SDISC  1230     // S - C
#define SINKN  4
#define TOPKN  405      // HB - SINK
#define CPAD   832      // 52 * 16
#define SDPAD  1232     // 77 * 16
#define EPSV   6.103515625e-03f

#if defined(__has_builtin)
#  if __has_builtin(__builtin_amdgcn_tensor_load_to_lds) && __has_builtin(__builtin_amdgcn_s_wait_tensorcnt)
#    define HAS_TDM 1
#  endif
#endif
#ifndef HAS_TDM
#  define HAS_TDM 0
#endif

static __device__ inline u16 f2bf_bits(float x) {
  unsigned int u = __float_as_uint(x);
  unsigned int r = (u + 0x7FFFu + ((u >> 16) & 1u)) >> 16;
  return (u16)r;
}
static __device__ inline u16 f2h_bits(float x) {
  __half h = __float2half(x);
  return *reinterpret_cast<u16*>(&h);
}

#if HAS_TDM
// TDM: copy a (tile_h x 32) bf16 tile (row stride = stride_elems) from global to LDS,
// padding each 64B row with 16B -> LDS row stride of 40 halves (matches fragment loader).
static __device__ inline void tdm_load_tile_a(const u16* gsrc, unsigned lds_off,
                                              int tensor_w, int tensor_h,
                                              int stride_elems, int tile_h) {
  unsigned long long ga = (unsigned long long)(const void*)gsrc;
  u32x4 g0;
  g0[0] = 1u;                                              // count=1, user mode
  g0[1] = lds_off;                                         // LDS byte address
  g0[2] = (unsigned)(ga & 0xFFFFFFFFu);                    // global addr [31:0]
  g0[3] = (unsigned)((ga >> 32) & 0x01FFFFFFu) | (2u << 30); // addr[56:32] | type=2
  i32x8 g1;
  g1[0] = (1 << 16)        // data_size = 2 bytes
        | (1 << 20)        // pad_enable
        | (3 << 22)        // pad_interval: 16 DWORDs (64B) between pads
        | (3 << 25);       // pad_amount: 4 DWORDs (16B)
  g1[1] = (tensor_w & 0xFFFF) << 16;                              // tensor_dim0[15:0]
  g1[2] = ((tensor_w >> 16) & 0xFFFF) | ((tensor_h & 0xFFFF) << 16); // dim0[31:16]|dim1[15:0]
  g1[3] = ((tensor_h >> 16) & 0xFFFF) | (32 << 16);               // dim1[31:16]|tile_dim0=32
  g1[4] = (tile_h & 0xFFFF);                                      // tile_dim1 | tile_dim2=0
  g1[5] = stride_elems;                                           // tensor_dim0_stride[31:0]
  g1[6] = 0;
  g1[7] = 0;
  i32x4 z4;
  z4[0] = 0; z4[1] = 0; z4[2] = 0; z4[3] = 0;
#if __clang_major__ >= 23
  i32x8 z8;
  z8[0] = 0; z8[1] = 0; z8[2] = 0; z8[3] = 0; z8[4] = 0; z8[5] = 0; z8[6] = 0; z8[7] = 0;
  __builtin_amdgcn_tensor_load_to_lds(g0, g1, z4, z4, z8, 0);
#else
  __builtin_amdgcn_tensor_load_to_lds(g0, g1, z4, z4, 0);
#endif
}
#endif

// ---------------------------------------------------------------- convert
__global__ __launch_bounds__(256) void cvt_bf16_kernel(const float* __restrict__ src,
                                                       u16* __restrict__ dst, int n) {
  for (int i = blockIdx.x * blockDim.x + threadIdx.x; i < n; i += gridDim.x * blockDim.x)
    dst[i] = f2bf_bits(src[i]);
}

// ---------------------------------------------------------------- GEMM (bf16 x bf16 -> f32)
// C[M][N] = A[M][K] * B[K][N], row-major, M,N % 128 == 0, K % 32 == 0.
// Double-buffered LDS; A tile moved by the Tensor Data Mover when available.
__global__ __launch_bounds__(256) void gemm_bf16_kernel(
    const u16* __restrict__ A, const u16* __restrict__ B, float* __restrict__ C,
    int M, int N, int K, int lda, int ldb, int ldc) {
  (void)N;
  __shared__ u16 As[2][128 * 40];
  __shared__ u16 Bs[2][128 * 40];
  const int tid = threadIdx.x;
  const int wid = tid >> 5;
  const int lane = tid & 31;
  const int bm = blockIdx.y * 128;
  const int bn = blockIdx.x * 128;
  const int wm = (wid >> 1) * 32;   // 4 waves along M
  const int wn = (wid & 1) * 64;    // 2 waves along N
  const int cn  = lane & 15;
  const int akc = (lane < 16) ? 0 : 16;   // A-fragment K offset (halves)
  const int bkc = (lane < 16) ? 0 : 16;   // B-fragment K offset (halves)

  v8f acc[2][4];
#pragma unroll
  for (int i = 0; i < 2; ++i)
#pragma unroll
    for (int j = 0; j < 4; ++j)
#pragma unroll
      for (int r = 0; r < 8; ++r) acc[i][j][r] = 0.f;

  auto load_a = [&](int buf, int kk) {
#if HAS_TDM
    if (wid == 0)
      tdm_load_tile_a(A + (size_t)bm * lda + kk,
                      (unsigned)(unsigned long long)(&As[buf][0]),
                      K - kk, M - bm, lda, 128);
#else
    int row = tid >> 1, cp = (tid & 1) * 16;
    const u16* src = A + (size_t)(bm + row) * lda + kk + cp;
    *(uint4*)(&As[buf][row * 40 + cp])     = *(const uint4*)(src);
    *(uint4*)(&As[buf][row * 40 + cp + 8]) = *(const uint4*)(src + 8);
    if (kk + 32 < K) __builtin_prefetch(src + 32, 0, 3);
#endif
  };
  auto load_b = [&](int buf, int kk) {
    int krow = tid >> 3, ncp = (tid & 7) * 16;
    const u16* src = B + (size_t)(kk + krow) * ldb + bn + ncp;
    u16 tmp[16];
    *(uint4*)(tmp)     = *(const uint4*)(src);
    *(uint4*)(tmp + 8) = *(const uint4*)(src + 8);
#pragma unroll
    for (int i = 0; i < 16; ++i) Bs[buf][(ncp + i) * 40 + krow] = tmp[i];
    if (kk + 32 < K) __builtin_prefetch(src + (size_t)32 * ldb, 0, 3);
  };

  load_a(0, 0);
  load_b(0, 0);
#if HAS_TDM
  if (wid == 0) __builtin_amdgcn_s_wait_tensorcnt(0);
#endif
  __syncthreads();

  const int nkt = K / 32;
  for (int kt = 0; kt < nkt; ++kt) {
    const int cur = kt & 1;
    if (kt + 1 < nkt) {
      load_a(cur ^ 1, (kt + 1) * 32);
      load_b(cur ^ 1, (kt + 1) * 32);
    }
    Frag a[2], b[4];
#pragma unroll
    for (int i = 0; i < 2; ++i) {
      int row = wm + i * 16 + cn;
      a[i].u[0] = *(const uint4*)(&As[cur][row * 40 + akc]);
      a[i].u[1] = *(const uint4*)(&As[cur][row * 40 + akc + 8]);
    }
#pragma unroll
    for (int j = 0; j < 4; ++j) {
      int col = wn + j * 16 + cn;
      b[j].u[0] = *(const uint4*)(&Bs[cur][col * 40 + bkc]);
      b[j].u[1] = *(const uint4*)(&Bs[cur][col * 40 + bkc + 8]);
    }
#pragma unroll
    for (int i = 0; i < 2; ++i)
#pragma unroll
      for (int j = 0; j < 4; ++j)
        acc[i][j] = __builtin_amdgcn_wmma_f32_16x16x32_bf16(
            false, a[i].bf, false, b[j].bf, (short)0, acc[i][j], false, false);
#if HAS_TDM
    if (wid == 0) __builtin_amdgcn_s_wait_tensorcnt(0);
#endif
    __syncthreads();
  }

  const int hi8 = (lane < 16) ? 0 : 8;
#pragma unroll
  for (int i = 0; i < 2; ++i)
#pragma unroll
    for (int j = 0; j < 4; ++j) {
      int row0 = bm + wm + i * 16 + hi8;
      int col  = bn + wn + j * 16 + cn;
#pragma unroll
      for (int r = 0; r < 8; ++r)
        C[(size_t)(row0 + r) * ldc + col] = acc[i][j][r];
    }
}

// ---------------------------------------------------------------- RoPE (in-place f32 + bf16 copies)
__global__ __launch_bounds__(256) void rope_kernel(float* __restrict__ q, float* __restrict__ k,
                                                   u16* __restrict__ qb, u16* __restrict__ kb) {
  int idx = blockIdx.x * blockDim.x + threadIdx.x;   // S*H*64 threads
  int i = idx & 63;
  int h = (idx >> 6) & (NHEAD - 1);
  int s = idx >> 10;
  float f = (float)s * __powf(10000.f, -2.f * (float)i / 128.f);
  float sn, cs;
  __sincosf(f, &sn, &cs);
  size_t base = (size_t)s * HIDDEN + h * DHEAD;
  {
    float x0 = q[base + i], x1 = q[base + i + 64];
    float o0 = x0 * cs - x1 * sn, o1 = x1 * cs + x0 * sn;
    q[base + i] = o0; q[base + i + 64] = o1;
    qb[base + i] = f2bf_bits(o0); qb[base + i + 64] = f2bf_bits(o1);
  }
  {
    float x0 = k[base + i], x1 = k[base + i + 64];
    float o0 = x0 * cs - x1 * sn, o1 = x1 * cs + x0 * sn;
    k[base + i] = o0; k[base + i + 64] = o1;
    kb[base + i] = f2bf_bits(o0); kb[base + i + 64] = f2bf_bits(o1);
  }
}

// ---------------------------------------------------------------- Flash attention
// block = 128 threads (4 waves), grid = (SEQ/64, NHEAD). Wave w owns query rows q0..q0+15.
__global__ __launch_bounds__(128) void attn_fa_kernel(
    const u16* __restrict__ Qb, const u16* __restrict__ Kb, const u16* __restrict__ Vb,
    float* __restrict__ Ao, float* __restrict__ Pr) {
  __shared__ u16 Ks[64 * 136];
  __shared__ u16 Vs[128 * 72];
  __shared__ u16 Ps[4][16 * 72];
  const int h = blockIdx.y;
  const int qt = blockIdx.x;
  const int tid = threadIdx.x;
  const int wid = tid >> 5;
  const int lane = tid & 31;
  const int q0 = qt * 64 + wid * 16;
  const int cn = lane & 15;
  const int hi8 = (lane < 16) ? 0 : 8;
  const int akc = (lane < 16) ? 0 : 16;
  const int bkc = (lane < 16) ? 0 : 16;
  const float scale = 0.0883883476483184f;  // 1/sqrt(128)

  Frag qf[4];
  {
    const u16* src = Qb + (size_t)(q0 + cn) * HIDDEN + h * DHEAD;
#pragma unroll
    for (int c = 0; c < 4; ++c) {
      qf[c].u[0] = *(const uint4*)(src + c * 32 + akc);
      qf[c].u[1] = *(const uint4*)(src + c * 32 + akc + 8);
    }
  }
  v8f Of[8];
  float mrow[8], lrow[8];
#pragma unroll
  for (int nf = 0; nf < 8; ++nf)
#pragma unroll
    for (int r = 0; r < 8; ++r) Of[nf][r] = 0.f;
#pragma unroll
  for (int r = 0; r < 8; ++r) { mrow[r] = -3.0e38f; lrow[r] = 0.f; }

  const int nkb = qt + 1;
  for (int kb = 0; kb < nkb; ++kb) {
    const int kk0 = kb * 64;
    {
      int row = tid >> 1, cp = (tid & 1) * 64;
      const u16* src = Kb + (size_t)(kk0 + row) * HIDDEN + h * DHEAD + cp;
#pragma unroll
      for (int i = 0; i < 8; ++i)
        *(uint4*)(&Ks[row * 136 + cp + i * 8]) = *(const uint4*)(src + i * 8);
    }
    {
      int row = tid >> 1, cp = (tid & 1) * 64;
      const u16* src = Vb + (size_t)(kk0 + row) * HIDDEN + h * DHEAD + cp;
#pragma unroll
      for (int i = 0; i < 64; ++i) Vs[(cp + i) * 72 + row] = src[i];
    }
    __syncthreads();

    v8f sc[4];
#pragma unroll
    for (int kn = 0; kn < 4; ++kn) {
      v8f acc;
#pragma unroll
      for (int r = 0; r < 8; ++r) acc[r] = 0.f;
      const u16* kr = &Ks[(kn * 16 + cn) * 136];
#pragma unroll
      for (int c = 0; c < 4; ++c) {
        Frag bfr;
        bfr.u[0] = *(const uint4*)(kr + c * 32 + bkc);
        bfr.u[1] = *(const uint4*)(kr + c * 32 + bkc + 8);
        acc = __builtin_amdgcn_wmma_f32_16x16x32_bf16(
            false, qf[c].bf, false, bfr.bf, (short)0, acc, false, false);
      }
      sc[kn] = acc;
    }
    float bmax[8];
#pragma unroll
    for (int r = 0; r < 8; ++r) bmax[r] = -3.0e38f;
#pragma unroll
    for (int kn = 0; kn < 4; ++kn) {
      int keyg = kk0 + kn * 16 + cn;
#pragma unroll
      for (int r = 0; r < 8; ++r) {
        float s = sc[kn][r] * scale;
        if (keyg > q0 + r + hi8) s = -3.0e38f;
        sc[kn][r] = s;
        bmax[r] = fmaxf(bmax[r], s);
      }
    }
#pragma unroll
    for (int off = 8; off >= 1; off >>= 1)
#pragma unroll
      for (int r = 0; r < 8; ++r) bmax[r] = fmaxf(bmax[r], __shfl_xor(bmax[r], off, 32));
    float alpha[8];
#pragma unroll
    for (int r = 0; r < 8; ++r) {
      float mn = fmaxf(mrow[r], bmax[r]);
      alpha[r] = __expf(mrow[r] - mn);
      mrow[r] = mn;
    }
    float psum[8];
#pragma unroll
    for (int r = 0; r < 8; ++r) psum[r] = 0.f;
#pragma unroll
    for (int kn = 0; kn < 4; ++kn)
#pragma unroll
      for (int r = 0; r < 8; ++r) {
        float p = __expf(sc[kn][r] - mrow[r]);
        sc[kn][r] = p;
        psum[r] += p;
      }
#pragma unroll
    for (int off = 8; off >= 1; off >>= 1)
#pragma unroll
      for (int r = 0; r < 8; ++r) psum[r] += __shfl_xor(psum[r], off, 32);
#pragma unroll
    for (int r = 0; r < 8; ++r) lrow[r] = lrow[r] * alpha[r] + psum[r];
#pragma unroll
    for (int nf = 0; nf < 8; ++nf)
#pragma unroll
      for (int r = 0; r < 8; ++r) Of[nf][r] *= alpha[r];
#pragma unroll
    for (int kn = 0; kn < 4; ++kn)
#pragma unroll
      for (int r = 0; r < 8; ++r)
        Ps[wid][(r + hi8) * 72 + kn * 16 + cn] = f2bf_bits(sc[kn][r]);
    __syncthreads();

    Frag pf[2];
#pragma unroll
    for (int c2 = 0; c2 < 2; ++c2) {
      pf[c2].u[0] = *(const uint4*)(&Ps[wid][cn * 72 + c2 * 32 + akc]);
      pf[c2].u[1] = *(const uint4*)(&Ps[wid][cn * 72 + c2 * 32 + akc + 8]);
    }
#pragma unroll
    for (int nf = 0; nf < 8; ++nf) {
      const u16* vr = &Vs[(nf * 16 + cn) * 72];
#pragma unroll
      for (int c2 = 0; c2 < 2; ++c2) {
        Frag vf;
        vf.u[0] = *(const uint4*)(vr + c2 * 32 + bkc);
        vf.u[1] = *(const uint4*)(vr + c2 * 32 + bkc + 8);
        Of[nf] = __builtin_amdgcn_wmma_f32_16x16x32_bf16(
            false, pf[c2].bf, false, vf.bf, (short)0, Of[nf], false, false);
      }
    }
    __syncthreads();
  }

  float invl[8];
#pragma unroll
  for (int r = 0; r < 8; ++r) invl[r] = 1.f / lrow[r];
#pragma unroll
  for (int nf = 0; nf < 8; ++nf)
#pragma unroll
    for (int r = 0; r < 8; ++r)
      Ao[(size_t)(q0 + r + hi8) * HIDDEN + h * DHEAD + nf * 16 + cn] = Of[nf][r] * invl[r];

  // Pass 2: normalized probs for the last R query rows (block-uniform condition).
  if (qt * 64 + 63 >= SEQ - RREC) {
    for (int kb = 0; kb < nkb; ++kb) {
      const int kk0 = kb * 64;
      {
        int row = tid >> 1, cp = (tid & 1) * 64;
        const u16* src = Kb + (size_t)(kk0 + row) * HIDDEN + h * DHEAD + cp;
#pragma unroll
        for (int i = 0; i < 8; ++i)
          *(uint4*)(&Ks[row * 136 + cp + i * 8]) = *(const uint4*)(src + i * 8);
      }
      __syncthreads();
#pragma unroll
      for (int kn = 0; kn < 4; ++kn) {
        v8f acc;
#pragma unroll
        for (int r = 0; r < 8; ++r) acc[r] = 0.f;
        const u16* kr = &Ks[(kn * 16 + cn) * 136];
#pragma unroll
        for (int c = 0; c < 4; ++c) {
          Frag bfr;
          bfr.u[0] = *(const uint4*)(kr + c * 32 + bkc);
          bfr.u[1] = *(const uint4*)(kr + c * 32 + bkc + 8);
          acc = __builtin_amdgcn_wmma_f32_16x16x32_bf16(
              false, qf[c].bf, false, bfr.bf, (short)0, acc, false, false);
        }
        int keyg = kk0 + kn * 16 + cn;
#pragma unroll
        for (int r = 0; r < 8; ++r) {
          int qg = q0 + r + hi8;
          int rr = qg - (SEQ - RREC);
          if (rr >= 0) {
            float p = (keyg > qg) ? 0.f : __expf(acc[r] * scale - mrow[r]) * invl[r];
            Pr[((size_t)h * RREC + rr) * SEQ + keyg] = p;
          }
        }
      }
      __syncthreads();
    }
  }
}

// ---------------------------------------------------------------- KV select (per head)
__global__ __launch_bounds__(256) void kv_select_kernel(
    const float* __restrict__ Pr, int* __restrict__ keep_idx, int* __restrict__ disc_idx,
    float* __restrict__ Wkeep, float* __restrict__ WdiscE) {
  __shared__ float sel[SEQ];
  __shared__ float colmean[SEQ];
  __shared__ unsigned char keep[SEQ];
  __shared__ int kidx[CKEEP];
  __shared__ int didx[SDISC];
  const int h = blockIdx.x;
  const int tid = threadIdx.x;
  const float* P = Pr + (size_t)h * RREC * SEQ;

  for (int s = tid; s < SEQ; s += 256) {
    float acc = 0.f, accm = 0.f;
    for (int r = 0; r < RREC; ++r) {
      float p = P[(size_t)r * SEQ + s];
      acc += p * (1.f - 0.5f * (float)r / (float)(RREC - 1));
      accm += p;
    }
    sel[s] = acc;
    colmean[s] = accm * (1.f / (float)RREC);
  }
  __syncthreads();
  for (int s = tid; s < SEQ; s += 256) {
    unsigned char kp;
    if (s < SINKN || s >= SEQ - RREC) kp = 1;
    else {
      float v = sel[s];
      int rank = 0;
      for (int j = SINKN; j < SEQ - RREC; ++j) {
        float vj = sel[j];
        rank += (vj > v) || (vj == v && j < s);
      }
      kp = (rank < TOPKN) ? 1 : 0;
    }
    keep[s] = kp;
  }
  __syncthreads();
  if (tid == 0) {
    int ck = 0, cd = 0;
    for (int s = 0; s < SEQ; ++s) {
      if (keep[s]) { if (ck < CKEEP) kidx[ck++] = s; }
      else         { if (cd < SDISC) didx[cd++] = s; }
    }
  }
  __syncthreads();
  for (int c = tid; c < CKEEP; c += 256) {
    int s = kidx[c];
    keep_idx[h * CKEEP + c] = s;
    float w = colmean[s];
    if (c >= CKEEP - RREC) {
      int i = c - (CKEEP - RREC);
      w *= (float)RREC / (float)(RREC - i);
    }
    Wkeep[h * CKEEP + c] = w + EPSV;
  }
  for (int d = tid; d < SDISC; d += 256) {
    int s = didx[d];
    disc_idx[h * SDISC + d] = s;
    WdiscE[h * SDISC + d] = colmean[s] + EPSV;
  }
}

// ---------------------------------------------------------------- gather + L2-normalize -> f16
// grid = (CPAD + SDPAD, NHEAD), block = 128
__global__ __launch_bounds__(128) void kv_norm16_kernel(
    const float* __restrict__ Kf, const int* __restrict__ keep_idx, const int* __restrict__ disc_idx,
    u16* __restrict__ nk, u16* __restrict__ dk) {
  __shared__ float red[128];
  const int h = blockIdx.y;
  const int row = blockIdx.x;
  const bool isKeep = row < CPAD;
  const int r = isKeep ? row : row - CPAD;
  const int lim = isKeep ? CKEEP : SDISC;
  u16* dst = isKeep ? (nk + ((size_t)h * CPAD + r) * DHEAD)
                    : (dk + ((size_t)h * SDPAD + r) * DHEAD);
  const int tid = threadIdx.x;
  if (r >= lim) { dst[tid] = 0; return; }
  const int s = isKeep ? keep_idx[h * CKEEP + r] : disc_idx[h * SDISC + r];
  float x = Kf[(size_t)s * HIDDEN + h * DHEAD + tid];
  red[tid] = x * x;
  __syncthreads();
  for (int off = 64; off >= 1; off >>= 1) {
    if (tid < off) red[tid] += red[tid + off];
    __syncthreads();
  }
  dst[tid] = f2h_bits(x * rsqrtf(red[0]));
}

// ---------------------------------------------------------------- similarity GEMM + argmax (f16 WMMA)
// grid = (SDPAD/16, NHEAD), block = 32 (one wave handles one 16-row disc tile vs all kept cols)
__global__ __launch_bounds__(32) void kv_sim_kernel(
    const u16* __restrict__ nk, const u16* __restrict__ dk,
    float* __restrict__ maxsim, int* __restrict__ maxidx) {
  const int h = blockIdx.y;
  const int dt = blockIdx.x;
  const int lane = threadIdx.x;
  const int cn = lane & 15;
  const int hi8 = (lane < 16) ? 0 : 8;
  const int akc = (lane < 16) ? 0 : 16;
  const int bkc = (lane < 16) ? 0 : 16;
  const u16* A = dk + ((size_t)h * SDPAD + dt * 16) * DHEAD;
  const u16* Bm = nk + (size_t)h * CPAD * DHEAD;

  Frag a[4];
#pragma unroll
  for (int c = 0; c < 4; ++c) {
    a[c].u[0] = *(const uint4*)(A + (size_t)cn * DHEAD + c * 32 + akc);
    a[c].u[1] = *(const uint4*)(A + (size_t)cn * DHEAD + c * 32 + akc + 8);
  }
  float best[8];
  int bcol[8];
#pragma unroll
  for (int r = 0; r < 8; ++r) { best[r] = -3.0e38f; bcol[r] = CKEEP - 1; }

  for (int ct = 0; ct < CPAD / 16; ++ct) {
    v8f acc;
#pragma unroll
    for (int r = 0; r < 8; ++r) acc[r] = 0.f;
    const u16* br = Bm + (size_t)(ct * 16 + cn) * DHEAD;
#pragma unroll
    for (int c = 0; c < 4; ++c) {
      Frag b;
      b.u[0] = *(const uint4*)(br + c * 32 + bkc);
      b.u[1] = *(const uint4*)(br + c * 32 + bkc + 8);
      acc = __builtin_amdgcn_wmma_f32_16x16x32_f16(
          false, a[c].h, false, b.h, (short)0, acc, false, false);
    }
    int col = ct * 16 + cn;
#pragma unroll
    for (int r = 0; r < 8; ++r) {
      float v = acc[r];
      if (v > best[r] || (v == best[r] && col < bcol[r])) { best[r] = v; bcol[r] = col; }
    }
  }
#pragma unroll
  for (int off = 8; off >= 1; off >>= 1)
#pragma unroll
    for (int r = 0; r < 8; ++r) {
      float ov = __shfl_xor(best[r], off, 32);
      int oc = __shfl_xor(bcol[r], off, 32);
      if (ov > best[r] || (ov == best[r] && oc < bcol[r])) { best[r] = ov; bcol[r] = oc; }
    }
  if (cn == 0) {
#pragma unroll
    for (int r = 0; r < 8; ++r) {
      int dj = dt * 16 + r + hi8;
      if (dj < SDISC) {
        maxsim[h * SDISC + dj] = best[r];
        int t = bcol[r];
        maxidx[h * SDISC + dj] = (t < CKEEP) ? t : (CKEEP - 1);
      }
    }
  }
}

// ---------------------------------------------------------------- init accumulators with kept rows
__global__ __launch_bounds__(128) void kv_init_kernel(
    const float* __restrict__ Kf, const float* __restrict__ Vf,
    const int* __restrict__ keep_idx, const float* __restrict__ Wkeep,
    float* __restrict__ ak, float* __restrict__ av, float* __restrict__ Wtot) {
  const int h = blockIdx.y, c = blockIdx.x, tid = threadIdx.x;
  const int s = keep_idx[h * CKEEP + c];
  const float w = Wkeep[h * CKEEP + c];
  const size_t dsti = ((size_t)h * CKEEP + c) * DHEAD + tid;
  const size_t srci = (size_t)s * HIDDEN + h * DHEAD + tid;
  ak[dsti] = Kf[srci] * w;
  av[dsti] = Vf[srci] * w;
  if (tid == 0) Wtot[h * CKEEP + c] = w;
}

// ---------------------------------------------------------------- scatter discarded rows
__global__ __launch_bounds__(128) void kv_scatter_kernel(
    const float* __restrict__ Kf, const float* __restrict__ Vf,
    const int* __restrict__ disc_idx, const float* __restrict__ WdiscE,
    const float* __restrict__ maxsim, const int* __restrict__ maxidx,
    float* __restrict__ ak, float* __restrict__ av, float* __restrict__ Wtot) {
  const int h = blockIdx.y, dj = blockIdx.x, tid = threadIdx.x;
  const float ms = maxsim[h * SDISC + dj];
  const float w = (ms >= 0.5f) ? WdiscE[h * SDISC + dj] : 0.f;
  if (w == 0.f) return;  // block-uniform
  const int t = maxidx[h * SDISC + dj];
  const int s = disc_idx[h * SDISC + dj];
  const size_t dsti = ((size_t)h * CKEEP + t) * DHEAD + tid;
  const size_t srci = (size_t)s * HIDDEN + h * DHEAD + tid;
  atomicAdd(&ak[dsti], Kf[srci] * w);
  atomicAdd(&av[dsti], Vf[srci] * w);
  if (tid == 0) atomicAdd(&Wtot[h * CKEEP + t], w);
}

// ---------------------------------------------------------------- normalize + f16 output
__global__ __launch_bounds__(128) void kv_finalize_kernel(
    const float* __restrict__ ak, const float* __restrict__ av, const float* __restrict__ Wtot,
    u16* __restrict__ outk, u16* __restrict__ outv) {
  const int h = blockIdx.y, c = blockIdx.x, tid = threadIdx.x;
  const float inv = 1.f / Wtot[h * CKEEP + c];
  const size_t i = ((size_t)h * CKEEP + c) * DHEAD + tid;
  outk[i] = f2h_bits(ak[i] * inv);
  outv[i] = f2h_bits(av[i] * inv);
}

// ================================================================ host
extern "C" void kernel_launch(void* const* d_in, const int* in_sizes, int n_in,
                              void* d_out, int out_size, void* d_ws, size_t ws_size,
                              hipStream_t stream) {
  (void)in_sizes; (void)n_in; (void)out_size; (void)ws_size;
  const float* hs = (const float*)d_in[0];
  const float* wq = (const float*)d_in[2];
  const float* wk = (const float*)d_in[3];
  const float* wv = (const float*)d_in[4];
  const float* wo = (const float*)d_in[5];

  const size_t NE = (size_t)SEQ * HIDDEN;  // 4,194,304
  char* ws = (char*)d_ws;
  size_t off = 0;
  auto alloc = [&](size_t bytes) -> char* {
    char* p = ws + off;
    off += (bytes + 255) & ~(size_t)255;
    return p;
  };
  u16* hb  = (u16*)alloc(NE * 2);
  u16* wqb = (u16*)alloc(NE * 2);
  u16* wkb = (u16*)alloc(NE * 2);
  u16* wvb = (u16*)alloc(NE * 2);
  u16* wob = (u16*)alloc(NE * 2);
  float* qf = (float*)alloc(NE * 4);
  float* kf = (float*)alloc(NE * 4);
  float* vf = (float*)alloc(NE * 4);
  u16* qb = (u16*)alloc(NE * 2);
  u16* kb = (u16*)alloc(NE * 2);
  u16* vb = (u16*)alloc(NE * 2);
  float* ao = (float*)alloc(NE * 4);
  u16* aob = (u16*)alloc(NE * 2);
  float* Pr = (float*)alloc((size_t)NHEAD * RREC * SEQ * 4);
  int* keep_idx = (int*)alloc((size_t)NHEAD * CKEEP * 4);
  int* disc_idx = (int*)alloc((size_t)NHEAD * SDISC * 4);
  float* Wkeep = (float*)alloc((size_t)NHEAD * CKEEP * 4);
  float* WdiscE = (float*)alloc((size_t)NHEAD * SDISC * 4);
  u16* nk = (u16*)alloc((size_t)NHEAD * CPAD * DHEAD * 2);
  u16* dk = (u16*)alloc((size_t)NHEAD * SDPAD * DHEAD * 2);
  float* maxsim = (float*)alloc((size_t)NHEAD * SDISC * 4);
  int* maxidx = (int*)alloc((size_t)NHEAD * SDISC * 4);
  float* ak = (float*)alloc((size_t)NHEAD * CKEEP * DHEAD * 4);
  float* av = (float*)alloc((size_t)NHEAD * CKEEP * DHEAD * 4);
  float* Wtot = (float*)alloc((size_t)NHEAD * CKEEP * 4);

  float* out = (float*)d_out;
  u16* outk = (u16*)(out + NE);
  u16* outv = outk + (size_t)NHEAD * CKEEP * DHEAD;

  // 1) f32 -> bf16 conversions
  cvt_bf16_kernel<<<4096, 256, 0, stream>>>(hs, hb, (int)NE);
  cvt_bf16_kernel<<<4096, 256, 0, stream>>>(wq, wqb, (int)NE);
  cvt_bf16_kernel<<<4096, 256, 0, stream>>>(wk, wkb, (int)NE);
  cvt_bf16_kernel<<<4096, 256, 0, stream>>>(wv, wvb, (int)NE);
  cvt_bf16_kernel<<<4096, 256, 0, stream>>>(wo, wob, (int)NE);

  // 2) QKV projections (WMMA bf16, TDM + double-buffered LDS)
  dim3 gg(HIDDEN / 128, SEQ / 128);
  gemm_bf16_kernel<<<gg, 256, 0, stream>>>(hb, wqb, qf, SEQ, HIDDEN, HIDDEN, HIDDEN, HIDDEN, HIDDEN);
  gemm_bf16_kernel<<<gg, 256, 0, stream>>>(hb, wkb, kf, SEQ, HIDDEN, HIDDEN, HIDDEN, HIDDEN, HIDDEN);
  gemm_bf16_kernel<<<gg, 256, 0, stream>>>(hb, wvb, vf, SEQ, HIDDEN, HIDDEN, HIDDEN, HIDDEN, HIDDEN);

  // 3) RoPE (+ bf16 Q/K), bf16 V
  rope_kernel<<<(SEQ * NHEAD * 64) / 256, 256, 0, stream>>>(qf, kf, qb, kb);
  cvt_bf16_kernel<<<4096, 256, 0, stream>>>(vf, vb, (int)NE);

  // 4) Flash attention (+ probs for last R rows)
  hipMemsetAsync(Pr, 0, (size_t)NHEAD * RREC * SEQ * 4, stream);
  attn_fa_kernel<<<dim3(SEQ / 64, NHEAD), 128, 0, stream>>>(qb, kb, vb, ao, Pr);

  // 5) Output projection
  cvt_bf16_kernel<<<4096, 256, 0, stream>>>(ao, aob, (int)NE);
  gemm_bf16_kernel<<<gg, 256, 0, stream>>>(aob, wob, out, SEQ, HIDDEN, HIDDEN, HIDDEN, HIDDEN, HIDDEN);

  // 6) KV-cache compaction
  kv_select_kernel<<<NHEAD, 256, 0, stream>>>(Pr, keep_idx, disc_idx, Wkeep, WdiscE);
  kv_norm16_kernel<<<dim3(CPAD + SDPAD, NHEAD), 128, 0, stream>>>(kf, keep_idx, disc_idx, nk, dk);
  kv_sim_kernel<<<dim3(SDPAD / 16, NHEAD), 32, 0, stream>>>(nk, dk, maxsim, maxidx);
  kv_init_kernel<<<dim3(CKEEP, NHEAD), 128, 0, stream>>>(kf, vf, keep_idx, Wkeep, ak, av, Wtot);
  kv_scatter_kernel<<<dim3(SDISC, NHEAD), 128, 0, stream>>>(kf, vf, disc_idx, WdiscE, maxsim, maxidx, ak, av, Wtot);
  kv_finalize_kernel<<<dim3(CKEEP, NHEAD), 128, 0, stream>>>(ak, av, Wtot, outk, outv);
}
